// RNNModel_76158360093034
// MI455X (gfx1250) — matrix operand order repaired and checked by reading, not compiled, not run on hardware
//
#include <hip/hip_runtime.h>
#include <stdint.h>

// Problem constants (match the reference harness).
#define N      8192
#define OBSD   64
#define ACT    16
#define ITERS  64

typedef __attribute__((ext_vector_type(16))) __bf16 v16bf;
typedef __attribute__((ext_vector_type(8)))  float  v8f;

// float -> bf16, round-to-nearest-even
__device__ __forceinline__ uint16_t f2bf(float f) {
  union { float f; uint32_t u; } x; x.f = f;
  uint32_t u = x.u;
  return (uint16_t)((u + 0x7FFFu + ((u >> 16) & 1u)) >> 16);
}

// ---------------------------------------------------------------------------
// One-time weight conversion: f32 [N*N] -> bf16 [N*N]. 8 elements per thread.
// 256 MB read + 128 MB write, amortized over 64 GEMV iterations.
// ---------------------------------------------------------------------------
__global__ void __launch_bounds__(256)
convert_w_kernel(const float* __restrict__ W, uint16_t* __restrict__ Wb) {
  size_t idx = (size_t)blockIdx.x * 256 + threadIdx.x;   // one uint4 (8 bf16)
  const float4* src = (const float4*)W;
  float4 a = src[idx * 2 + 0];
  float4 b = src[idx * 2 + 1];
  uint4 o;
  o.x = (uint32_t)f2bf(a.x) | ((uint32_t)f2bf(a.y) << 16);
  o.y = (uint32_t)f2bf(a.z) | ((uint32_t)f2bf(a.w) << 16);
  o.z = (uint32_t)f2bf(b.x) | ((uint32_t)f2bf(b.y) << 16);
  o.w = (uint32_t)f2bf(b.z) | ((uint32_t)f2bf(b.w) << 16);
  ((uint4*)Wb)[idx] = o;
}

// ---------------------------------------------------------------------------
// Build hin = [1, obs, h[65:]] as bf16. src is f32 (rnn_state on iter 0,
// the h workspace afterwards).
// ---------------------------------------------------------------------------
__global__ void __launch_bounds__(256)
build_hin_kernel(const float* __restrict__ src, const float* __restrict__ obs,
                 uint16_t* __restrict__ hin) {
  int i = blockIdx.x * 256 + threadIdx.x;                // N total
  float v = (i == 0) ? 1.0f : ((i <= OBSD) ? obs[i - 1] : src[i]);
  hin[i] = f2bf(v);
}

// ---------------------------------------------------------------------------
// h_out = tanh(Wb @ hin).  Grid: N/16 blocks x 256 threads (8 wave32).
// Block b owns rows [16b, 16b+16). Wave w accumulates K-slice
// [1024w, 1024w+1024) via 32 x v_wmma_f32_16x16x32_bf16, with B = hin
// broadcast into all 16 columns (column 0 extracted afterwards).
// A-fragment layout (16-bit A 16x32): lanes 0-15 hold K {0..7,16..23},
// lanes 16-31 hold K {8..15,24..31}, row = m_base + (lane&15).
// B-fragment layout (16-bit B 32x16): lanes 0-15 hold K {0..7,16..23},
// lanes 16-31 hold K {8..15,24..31} (column = lane&15; identical columns here).
// ---------------------------------------------------------------------------
__global__ void __launch_bounds__(256)
gemv_tanh_kernel(const uint16_t* __restrict__ Wb,
                 const uint16_t* __restrict__ hin,
                 float* __restrict__ hout) {
  __shared__ uint4 sh_hin[N / 8];        // 16 KB: whole hin vector in LDS
  __shared__ float red[8 * 16];          // per-wave partial row sums

  const int tid = threadIdx.x;

  // Stage hin into LDS cooperatively (all 256 threads, uniform -> EXEC all 1s)
  for (int i = tid; i < N / 8; i += 256) sh_hin[i] = ((const uint4*)hin)[i];
  __syncthreads();

  const int wave  = tid >> 5;
  const int lid   = tid & 31;
  const int hi    = lid >> 4;                       // lane half (0/1)
  const int row   = blockIdx.x * 16 + (lid & 15);
  const uint32_t* wrow = (const uint32_t*)Wb + (size_t)row * (N / 2);
  const int kbase = wave * (N / 8);                 // 1024-element K slice

  union Frag { uint4 q[2]; v16bf v; };

  v8f acc = {};

#pragma unroll 4
  for (int kc = 0; kc < 32; ++kc) {
    const int k0 = kbase + kc * 32;

    // A tile: two 16-byte loads per lane per the documented layout
    Frag a;
    const int ka = k0 + hi * 8;
    a.q[0] = *(const uint4*)(wrow + (ka >> 1));
    a.q[1] = *(const uint4*)(wrow + ((ka + 16) >> 1));

    // Keep the W stream ahead in cache (global_prefetch_b8)
    if ((kc & 7) == 0)
      __builtin_prefetch((const void*)(wrow + (ka >> 1) + 128), 0, 0);

    // B tile from LDS: same K order as A (K {0..7,16..23} / {8..15,24..31})
    Frag b;
    const int kb = k0 + hi * 8;
    b.q[0] = sh_hin[(kb >> 3) + 0];
    b.q[1] = sh_hin[(kb >> 3) + 2];

    acc = __builtin_amdgcn_wmma_f32_16x16x32_bf16(
        /*neg_a=*/false, a.v, /*neg_b=*/false, b.v,
        /*c_mod=*/(short)0, acc, /*reuse_a=*/false, /*reuse_b=*/false);
  }

  // Column 0 of C: lane 0 holds rows 0..7 (VGPR r -> M=r),
  // lane 16 holds rows 8..15 (VGPR r -> M=8+r).
  if (lid == 0 || lid == 16) {
    const int base = wave * 16 + hi * 8;
#pragma unroll
    for (int r = 0; r < 8; ++r) red[base + r] = acc[r];
  }
  __syncthreads();

  // Reduce across the 8 waves' K-slices, apply tanh, store f32 h.
  if (tid < 16) {
    float s = 0.0f;
#pragma unroll
    for (int w = 0; w < 8; ++w) s += red[w * 16 + tid];
    hout[blockIdx.x * 16 + tid] = tanhf(s);
  }
}

// ---------------------------------------------------------------------------
// Emit the last ACT elements of h as the policy action.
// ---------------------------------------------------------------------------
__global__ void writeout_kernel(const float* __restrict__ h,
                                float* __restrict__ out) {
  int i = threadIdx.x;
  out[i] = h[N - ACT + i];
}

extern "C" void kernel_launch(void* const* d_in, const int* in_sizes, int n_in,
                              void* d_out, int out_size, void* d_ws, size_t ws_size,
                              hipStream_t stream) {
  (void)in_sizes; (void)n_in; (void)out_size; (void)ws_size;

  const float* W   = (const float*)d_in[0];   // [N, N] f32
  const float* obs = (const float*)d_in[1];   // [OBSD] f32
  const float* h0  = (const float*)d_in[2];   // [N]    f32

  // Workspace layout: bf16 W (128 MB) | bf16 hin (16 KB) | f32 h (32 KB)
  char* ws = (char*)d_ws;
  uint16_t* Wb  = (uint16_t*)ws;
  uint16_t* hin = (uint16_t*)(ws + (size_t)N * N * 2);
  float*    h   = (float*)   (ws + (size_t)N * N * 2 + (size_t)N * 2);

  // One-time f32 -> bf16 conversion (W then lives in the 192 MB L2).
  convert_w_kernel<<<(unsigned)(((size_t)N * N / 8) / 256), 256, 0, stream>>>(W, Wb);

  const float* hsrc = h0;
  for (int it = 0; it < ITERS; ++it) {
    build_hin_kernel<<<N / 256, 256, 0, stream>>>(hsrc, obs, hin);
    gemv_tanh_kernel<<<N / 16, 256, 0, stream>>>(Wb, hin, h);
    hsrc = h;   // kernel boundary = global sync between iterations
  }

  writeout_kernel<<<1, ACT, 0, stream>>>(h, (float*)d_out);
}
